// RWKV7BlockInfctx_14869176778902
// MI455X (gfx1250) — matrix operand order repaired and checked by reading, not compiled, not run on hardware
//
#include <hip/hip_runtime.h>
#include <cstddef>

// ---------------- problem constants (RWKV7 block) ----------------
constexpr int Bq = 4, Tq = 1024, Cq = 2048, Hq = 32, Nq = 64;
constexpr int DFFq = 8192, DWq = 64, DVq = 32, DAq = 64, DGq = 128;
constexpr int MT = Bq * Tq;                 // 4096 token rows
constexpr size_t BTC = (size_t)MT * Cq;     // 8,388,608
constexpr float GN_EPS = 64e-5f;

typedef __attribute__((ext_vector_type(16))) __bf16 v16bf;
typedef __attribute__((ext_vector_type(8)))  float  v8f;

__device__ __forceinline__ float sigmoidf_(float x) { return 1.0f / (1.0f + expf(-x)); }

// =================================================================
// Elementwise fp32 -> bf16 with optional activation
// mode: 0=copy 1=tanh 2=sigmoid 3=relu^2
// =================================================================
__global__ void act_cvt_kernel(const float* __restrict__ in, __bf16* __restrict__ out,
                               size_t n, int mode) {
  size_t i = (size_t)blockIdx.x * blockDim.x + threadIdx.x;
  size_t stride = (size_t)gridDim.x * blockDim.x;
  for (; i < n; i += stride) {
    float v = in[i];
    if (mode == 1)      v = tanhf(v);
    else if (mode == 2) v = sigmoidf_(v);
    else if (mode == 3) { v = fmaxf(v, 0.0f); v = v * v; }
    out[i] = (__bf16)v;
  }
}

// =================================================================
// LayerNorm over C per row; one 256-thread block per row
// =================================================================
__global__ __launch_bounds__(256) void ln_kernel(const float* __restrict__ x,
                                                 const float* __restrict__ w,
                                                 const float* __restrict__ b,
                                                 float* __restrict__ out) {
  const int row = blockIdx.x, tid = threadIdx.x;
  const float* xr = x + (size_t)row * Cq;
  __shared__ float red[256];
  float s = 0.f;
  for (int c = tid; c < Cq; c += 256) s += xr[c];
  red[tid] = s; __syncthreads();
  for (int st = 128; st > 0; st >>= 1) { if (tid < st) red[tid] += red[tid + st]; __syncthreads(); }
  float mu = red[0] / Cq; __syncthreads();
  float v = 0.f;
  for (int c = tid; c < Cq; c += 256) { float d = xr[c] - mu; v += d * d; }
  red[tid] = v; __syncthreads();
  for (int st = 128; st > 0; st >>= 1) { if (tid < st) red[tid] += red[tid + st]; __syncthreads(); }
  float inv = rsqrtf(red[0] / Cq + 1e-5f);
  float* orow = out + (size_t)row * Cq;
  for (int c = tid; c < Cq; c += 256) orow[c] = (xr[c] - mu) * inv * w[c] + b[c];
}

// =================================================================
// Token-shift mix: 6 bf16 outputs (attention branch)
// =================================================================
__global__ void mix6_kernel(const float* __restrict__ xn, const float* __restrict__ shift,
                            const float* cr, const float* cw, const float* ck,
                            const float* cv, const float* ca, const float* cg,
                            __bf16* xr, __bf16* xw, __bf16* xk,
                            __bf16* xv, __bf16* xa, __bf16* xg) {
  size_t i = (size_t)blockIdx.x * blockDim.x + threadIdx.x;
  if (i >= BTC) return;
  int c = (int)(i % Cq);
  size_t bt = i / Cq;
  int t = (int)(bt % Tq); int b = (int)(bt / Tq);
  float cur = xn[i];
  float prev = (t == 0) ? shift[(size_t)b * Cq + c] : xn[i - Cq];
  float xx = prev - cur;
  xr[i] = (__bf16)(cur + xx * cr[c]);
  xw[i] = (__bf16)(cur + xx * cw[c]);
  xk[i] = (__bf16)(cur + xx * ck[c]);
  xv[i] = (__bf16)(cur + xx * cv[c]);
  xa[i] = (__bf16)(cur + xx * ca[c]);
  xg[i] = (__bf16)(cur + xx * cg[c]);
}

// Token-shift mix: single bf16 output (FFN branch)
__global__ void mix1_kernel(const float* __restrict__ xn, const float* __restrict__ shift,
                            const float* ck, __bf16* xk) {
  size_t i = (size_t)blockIdx.x * blockDim.x + threadIdx.x;
  if (i >= BTC) return;
  int c = (int)(i % Cq);
  size_t bt = i / Cq;
  int t = (int)(bt % Tq); int b = (int)(bt / Tq);
  float cur = xn[i];
  float prev = (t == 0) ? shift[(size_t)b * Cq + c] : xn[i - Cq];
  xk[i] = (__bf16)(cur + (prev - cur) * ck[c]);
}

// =================================================================
// bf16 WMMA GEMM:  C[M,Nd] = A[M,Kd] * B[Kd,Nd]  (fp32 accumulate)
// Block tile 128x64, BK=32, 256 threads = 8 waves (wave32).
// Each wave: 2x2 grid of v_wmma_f32_16x16x32_bf16 tiles (32x32).
// Double-buffered LDS; B staged transposed (N-major, padded) so both
// A and B fragments are 2x ds_load_b128 per 16x16 operand.
// Requires M % 128 == 0, Kd % 32 == 0. FULLN: Nd % 64 == 0.
// =================================================================
#define GBM 128
#define GBN 64
#define GBK 32
#define BSTR 40   // padded per-column stride of transposed B tile (bf16 units)

template <bool FULLN>
__global__ __launch_bounds__(256) void gemm_bf16_wmma(
    const __bf16* __restrict__ A, const __bf16* __restrict__ Bm,
    float* __restrict__ C, int M, int Nd, int Kd) {
  __shared__ __bf16 As[2][GBM * GBK];   // 2 x 8 KB
  __shared__ __bf16 Bs[2][GBN * BSTR];  // 2 x 5 KB (transposed + padded)

  const int tid  = threadIdx.x;
  const int wave = tid >> 5;
  const int lane = tid & 31;
  const int wm   = wave >> 1;     // 0..3 : 32-row strip
  const int wn   = wave & 1;      // 0..1 : 32-col strip
  const int half = lane >> 4;     // K-half selector
  const int l16  = lane & 15;

  const int blockM = blockIdx.x * GBM;
  const int blockN = blockIdx.y * GBN;

  v8f acc[2][2];
  #pragma unroll
  for (int i = 0; i < 2; ++i)
    #pragma unroll
    for (int j = 0; j < 2; ++j)
      #pragma unroll
      for (int e = 0; e < 8; ++e) acc[i][j][e] = 0.0f;

  // ---- tile loaders --------------------------------------------
  auto loadA = [&](int k0, int buf) {
    #pragma unroll
    for (int it = 0; it < 2; ++it) {
      int idx = tid + it * 256;           // 0..511 (4 uint4 per 32-elem row)
      int row = idx >> 2;
      int c4  = idx & 3;
      ((uint4*)As[buf])[idx] =
          *(const uint4*)(A + (size_t)(blockM + row) * Kd + k0 + c4 * 8);
    }
  };
  auto loadB = [&](int k0, int buf) {
    int row  = tid >> 3;                  // k within tile, 0..31
    int c4   = tid & 7;                   // 8-col group
    int gcol = blockN + c4 * 8;
    alignas(16) __bf16 tmp[8];
    if (FULLN || gcol + 8 <= Nd) {
      *(uint4*)tmp = *(const uint4*)(Bm + (size_t)(k0 + row) * Nd + gcol);
    } else {
      #pragma unroll
      for (int e = 0; e < 8; ++e)
        tmp[e] = (gcol + e < Nd) ? Bm[(size_t)(k0 + row) * Nd + gcol + e] : (__bf16)0.0f;
    }
    // transpose into LDS: Bs[col][k]
    #pragma unroll
    for (int e = 0; e < 8; ++e)
      Bs[buf][(c4 * 8 + e) * BSTR + row] = tmp[e];
  };

  const int ntiles = Kd / GBK;
  loadA(0, 0); loadB(0, 0);
  __syncthreads();

  union Frag { uint4 u[2]; v16bf v; };

  for (int kt = 0; kt < ntiles; ++kt) {
    const int buf = kt & 1;
    if (kt + 1 < ntiles) { loadA((kt + 1) * GBK, buf ^ 1); loadB((kt + 1) * GBK, buf ^ 1); }
    if (kt + 2 < ntiles)  // gfx1250 speculative prefetch (global_prefetch_b8)
      __builtin_prefetch(A + (size_t)(blockM + (tid >> 1)) * Kd + (kt + 2) * GBK, 0, 1);

    // ---- fragments: A lanes = M rows, K pairs packed; elements
    // [0..7] = ap[8*half .. +7], [8..15] = ap[16+8*half .. +7]
    v16bf afrag[2], bfrag[2];
    #pragma unroll
    for (int im = 0; im < 2; ++im) {
      const __bf16* ap = &As[buf][(wm * 32 + im * 16 + l16) * GBK];
      Frag f;
      f.u[0] = *(const uint4*)(ap + 8 * half);
      f.u[1] = *(const uint4*)(ap + 16 + 8 * half);
      afrag[im] = f.v;
    }
    // ---- B lanes = N cols; elements [0..7] = bp[8*half .. +7], [8..15] = bp[16+8*half .. +7]
    #pragma unroll
    for (int in = 0; in < 2; ++in) {
      const __bf16* bp = &Bs[buf][(wn * 32 + in * 16 + l16) * BSTR];
      Frag f;
      f.u[0] = ((const uint4*)bp)[half];
      f.u[1] = ((const uint4*)bp)[2 + half];
      bfrag[in] = f.v;
    }
    #pragma unroll
    for (int im = 0; im < 2; ++im)
      #pragma unroll
      for (int in = 0; in < 2; ++in)
        acc[im][in] = __builtin_amdgcn_wmma_f32_16x16x32_bf16(
            false, afrag[im], false, bfrag[in], (short)0, acc[im][in], false, false);
    __syncthreads();
  }

  // ---- epilogue: C/D layout (VGPR r -> M=r+8*half, N=lane&15)
  #pragma unroll
  for (int im = 0; im < 2; ++im)
    #pragma unroll
    for (int in = 0; in < 2; ++in)
      #pragma unroll
      for (int r = 0; r < 8; ++r) {
        int m = blockM + wm * 32 + im * 16 + r + 8 * half;
        int n = blockN + wn * 32 + in * 16 + l16;
        if (FULLN || n < Nd) C[(size_t)m * Nd + n] = acc[im][in][r];
      }
}

// =================================================================
// WKV prep: per-(b,t,h) head of 64 channels.
// Produces wh=exp(-exp(w)), kh=k*(1+(a-1)*k_a), vh, aa=-kk_norm, bb=kk_norm*a
// =================================================================
__global__ __launch_bounds__(64) void prep_wkv_kernel(
    const float* __restrict__ k, const float* __restrict__ wlin,
    const float* __restrict__ vraw, const float* __restrict__ vlin,
    const float* __restrict__ alin, const float* __restrict__ v_first,
    const float* __restrict__ w0, const float* __restrict__ v0,
    const float* __restrict__ a0, const float* __restrict__ k_k,
    const float* __restrict__ k_a,
    float* __restrict__ wh, float* __restrict__ kh, float* __restrict__ vh,
    float* __restrict__ aab, float* __restrict__ bbb) {
  const int row = blockIdx.x;            // b*T*H + t*H + h
  const int n   = threadIdx.x;           // 0..63
  const int h   = row % Hq;
  const size_t bt = row / Hq;
  const int c   = h * Nq + n;
  const size_t idx = bt * Cq + c;

  __shared__ float red[64];
  float kv  = k[idx];
  float kkv = kv * k_k[c];
  red[n] = kkv * kkv; __syncthreads();
  for (int s = 32; s > 0; s >>= 1) { if (n < s) red[n] += red[n + s]; __syncthreads(); }
  float nrm = fmaxf(sqrtf(red[0]), 1e-12f);
  float kkn = kkv / nrm;

  float a  = sigmoidf_(a0[c] + alin[idx]);
  float wv = w0[c] + wlin[idx];
  float w  = -log1pf(expf(-wv)) - 0.5f;       // -softplus(-wv) - 0.5
  float vr = vraw[idx];
  float vv = vr + (v_first[idx] - vr) * sigmoidf_(v0[c] + vlin[idx]);

  wh[idx]  = expf(-expf(w));
  kh[idx]  = kv * (1.0f + (a - 1.0f) * k_a[c]);
  vh[idx]  = vv;
  aab[idx] = -kkn;
  bbb[idx] = kkn * a;
}

// =================================================================
// Sequential WKV scan. One block per (b,h); 256 threads; thread owns
// row v = tid&63 and 16 k-columns [kg*16, kg*16+16). State in registers.
// =================================================================
__global__ __launch_bounds__(256) void wkv_scan_kernel(
    const float* __restrict__ r, const float* __restrict__ w,
    const float* __restrict__ k, const float* __restrict__ v,
    const float* __restrict__ aa, const float* __restrict__ bb,
    const float* __restrict__ S0, float* __restrict__ y,
    float* __restrict__ S_out) {
  const int bh = blockIdx.x;
  const int b  = bh / Hq, h = bh % Hq;
  const int tid  = threadIdx.x;
  const int vrow = tid & 63;
  const int kg   = tid >> 6;             // 0..3

  float S[16];
  const size_t sbase = ((size_t)bh) * Nq * Nq + (size_t)vrow * Nq + kg * 16;
  #pragma unroll
  for (int i = 0; i < 16; ++i) S[i] = S0[sbase + i];

  __shared__ float vec[6 * 64];          // r | w | k | v | aa | bb
  __shared__ float part[4][64];

  const size_t tbase = ((size_t)b * Tq) * Cq + (size_t)h * Nq;
  for (int t = 0; t < Tq; ++t) {
    const size_t o = tbase + (size_t)t * Cq;
    if (tid < 64) {
      vec[tid] = r[o + tid]; vec[64 + tid] = w[o + tid]; vec[128 + tid] = k[o + tid];
    } else if (tid < 128) {
      int n = tid - 64;
      vec[192 + n] = v[o + n]; vec[256 + n] = aa[o + n]; vec[320 + n] = bb[o + n];
    }
    __syncthreads();

    float p = 0.f;
    #pragma unroll
    for (int i = 0; i < 16; ++i) p += S[i] * vec[256 + kg * 16 + i];  // S . aa
    part[kg][vrow] = p; __syncthreads();
    float sa = part[0][vrow] + part[1][vrow] + part[2][vrow] + part[3][vrow];
    float vt = vec[192 + vrow];

    float q = 0.f;
    #pragma unroll
    for (int i = 0; i < 16; ++i) {
      int kk = kg * 16 + i;
      S[i] = S[i] * vec[64 + kk] + sa * vec[320 + kk] + vt * vec[128 + kk];
      q += S[i] * vec[kk];                                            // S . r
    }
    __syncthreads();
    part[kg][vrow] = q; __syncthreads();
    if (kg == 0) y[o + vrow] = part[0][vrow] + part[1][vrow] + part[2][vrow] + part[3][vrow];
    __syncthreads();
  }
  #pragma unroll
  for (int i = 0; i < 16; ++i) S_out[sbase + i] = S[i];
}

// =================================================================
// GroupNorm(H groups) + bonus residual + gate -> bf16
// =================================================================
__global__ __launch_bounds__(64) void gn_resid_gate_kernel(
    const float* __restrict__ y, const float* __restrict__ r,
    const float* __restrict__ kh, const float* __restrict__ vh,
    const float* __restrict__ g, const float* __restrict__ r_k,
    const float* __restrict__ lnx_w, const float* __restrict__ lnx_b,
    __bf16* __restrict__ og) {
  const int row = blockIdx.x;            // b*T*H + t*H + h
  const int n   = threadIdx.x;
  const int h   = row % Hq;
  const size_t bt = row / Hq;
  const int c   = h * Nq + n;
  const size_t idx = bt * Cq + c;

  __shared__ float red[64];
  float yv = y[idx];
  red[n] = yv; __syncthreads();
  for (int s = 32; s > 0; s >>= 1) { if (n < s) red[n] += red[n + s]; __syncthreads(); }
  float mu = red[0] / Nq; __syncthreads();
  float d = yv - mu;
  red[n] = d * d; __syncthreads();
  for (int s = 32; s > 0; s >>= 1) { if (n < s) red[n] += red[n + s]; __syncthreads(); }
  float var = red[0] / Nq; __syncthreads();
  red[n] = r[idx] * kh[idx] * r_k[c]; __syncthreads();
  for (int s = 32; s > 0; s >>= 1) { if (n < s) red[n] += red[n + s]; __syncthreads(); }
  float dot = red[0];

  float xg  = d * rsqrtf(var + GN_EPS) * lnx_w[c] + lnx_b[c];
  float out = xg + dot * vh[idx];
  og[idx] = (__bf16)(out * g[idx]);
}

// =================================================================
// out = a + b
// =================================================================
__global__ void add_kernel(const float* __restrict__ a, const float* __restrict__ b,
                           float* __restrict__ out, size_t n) {
  size_t i = (size_t)blockIdx.x * blockDim.x + threadIdx.x;
  size_t stride = (size_t)gridDim.x * blockDim.x;
  for (; i < n; i += stride) out[i] = a[i] + b[i];
}

// =================================================================
// Host-side orchestration
// =================================================================
extern "C" void kernel_launch(void* const* d_in, const int* in_sizes, int n_in,
                              void* d_out, int out_size, void* d_ws, size_t ws_size,
                              hipStream_t stream) {
  (void)in_sizes; (void)n_in; (void)out_size; (void)ws_size;

  const float* x         = (const float*)d_in[0];
  const float* v_first   = (const float*)d_in[1];
  const float* att_shift = (const float*)d_in[2];
  const float* ffn_shift = (const float*)d_in[3];
  const float* wkv_state = (const float*)d_in[4];
  const float* ln1_w = (const float*)d_in[5];  const float* ln1_b = (const float*)d_in[6];
  const float* ln2_w = (const float*)d_in[7];  const float* ln2_b = (const float*)d_in[8];
  const float* x_r = (const float*)d_in[9];  const float* x_w = (const float*)d_in[10];
  const float* x_k = (const float*)d_in[11]; const float* x_v = (const float*)d_in[12];
  const float* x_a = (const float*)d_in[13]; const float* x_g = (const float*)d_in[14];
  const float* Wr = (const float*)d_in[15]; const float* Wk = (const float*)d_in[16];
  const float* Wv = (const float*)d_in[17]; const float* Wo = (const float*)d_in[18];
  const float* w0 = (const float*)d_in[19]; const float* w1 = (const float*)d_in[20];
  const float* w2 = (const float*)d_in[21];
  const float* v0 = (const float*)d_in[22]; const float* v1 = (const float*)d_in[23];
  const float* v2 = (const float*)d_in[24];
  const float* a0 = (const float*)d_in[25]; const float* a1 = (const float*)d_in[26];
  const float* a2 = (const float*)d_in[27];
  const float* g1 = (const float*)d_in[28]; const float* g2 = (const float*)d_in[29];
  const float* k_k = (const float*)d_in[30]; const float* k_a = (const float*)d_in[31];
  const float* r_k = (const float*)d_in[32];
  const float* lnx_w = (const float*)d_in[33]; const float* lnx_b = (const float*)d_in[34];
  const float* ffn_xk = (const float*)d_in[35];
  const float* WkF = (const float*)d_in[36]; const float* WvF = (const float*)d_in[37];

  // ---- workspace allocator (256B aligned bump) ----
  size_t off = 0;
  auto alloc = [&](size_t bytes) -> char* {
    char* p = (char*)d_ws + off;
    off += (bytes + 255) & ~(size_t)255;
    return p;
  };
  const size_t CC = (size_t)Cq * Cq;

  __bf16* Wr_bf = (__bf16*)alloc(CC * 2);
  __bf16* Wk_bf = (__bf16*)alloc(CC * 2);
  __bf16* Wv_bf = (__bf16*)alloc(CC * 2);
  __bf16* Wo_bf = (__bf16*)alloc(CC * 2);
  __bf16* w1_bf = (__bf16*)alloc((size_t)Cq * DWq * 2);
  __bf16* w2_bf = (__bf16*)alloc((size_t)DWq * Cq * 2);
  __bf16* v1_bf = (__bf16*)alloc((size_t)Cq * DVq * 2);
  __bf16* v2_bf = (__bf16*)alloc((size_t)DVq * Cq * 2);
  __bf16* a1_bf = (__bf16*)alloc((size_t)Cq * DAq * 2);
  __bf16* a2_bf = (__bf16*)alloc((size_t)DAq * Cq * 2);
  __bf16* g1_bf = (__bf16*)alloc((size_t)Cq * DGq * 2);
  __bf16* g2_bf = (__bf16*)alloc((size_t)DGq * Cq * 2);
  __bf16* WkF_bf = (__bf16*)alloc((size_t)Cq * DFFq * 2);
  __bf16* WvF_bf = (__bf16*)alloc((size_t)DFFq * Cq * 2);

  float*  xn   = (float*)alloc(BTC * 4);
  __bf16* xr   = (__bf16*)alloc(BTC * 2);   // xr..xg contiguous; reused as kf later
  __bf16* xw   = (__bf16*)alloc(BTC * 2);
  __bf16* xkb  = (__bf16*)alloc(BTC * 2);
  __bf16* xv   = (__bf16*)alloc(BTC * 2);
  __bf16* xa   = (__bf16*)alloc(BTC * 2);
  __bf16* xg   = (__bf16*)alloc(BTC * 2);
  float*  rbuf = (float*)alloc(BTC * 4);
  float*  kbuf = (float*)alloc(BTC * 4);
  float*  vbuf = (float*)alloc(BTC * 4);
  float*  hw   = (float*)alloc((size_t)MT * DWq * 4);
  float*  hv   = (float*)alloc((size_t)MT * DVq * 4);
  float*  ha   = (float*)alloc((size_t)MT * DAq * 4);
  float*  hg   = (float*)alloc((size_t)MT * DGq * 4);
  __bf16* hw_bf = (__bf16*)alloc((size_t)MT * DWq * 2);
  __bf16* hv_bf = (__bf16*)alloc((size_t)MT * DVq * 2);
  __bf16* ha_bf = (__bf16*)alloc((size_t)MT * DAq * 2);
  __bf16* hg_bf = (__bf16*)alloc((size_t)MT * DGq * 2);
  float*  wlin = (float*)alloc(BTC * 4);
  float*  vlin = (float*)alloc(BTC * 4);
  float*  alin = (float*)alloc(BTC * 4);
  float*  gbuf = (float*)alloc(BTC * 4);
  float*  wh   = (float*)alloc(BTC * 4);
  float*  kh   = (float*)alloc(BTC * 4);
  float*  vh   = (float*)alloc(BTC * 4);
  float*  aab  = (float*)alloc(BTC * 4);
  float*  bbb  = (float*)alloc(BTC * 4);
  float*  ybuf = (float*)alloc(BTC * 4);
  __bf16* og   = (__bf16*)alloc(BTC * 2);
  float*  hffn = (float*)alloc((size_t)MT * DFFq * 4);
  // aliases (producer fully consumed before re-use)
  float*  att  = vlin;                      // attention projection output
  float*  x1   = alin;                      // residual after attention
  float*  xn2  = kbuf;                      // LN2 output
  __bf16* xkf  = og;                        // FFN mixed input (bf16)
  __bf16* kf   = xr;                        // relu^2 activation (bf16), fits in xr..xg
  float*  ffnout = vbuf;

  float* out_x = (float*)d_out;             // [B,T,C]
  float* out_S = out_x + BTC;               // [B,H,N,N]

  auto cvt = [&](const float* in, __bf16* out, size_t n, int mode) {
    int blocks = (int)((n + 255) / 256); if (blocks > 65535) blocks = 65535;
    act_cvt_kernel<<<blocks, 256, 0, stream>>>(in, out, n, mode);
  };
  auto gemm = [&](const __bf16* A, const __bf16* Bm, float* Cp, int M, int Nd, int Kd) {
    dim3 grid(M / GBM, (Nd + GBN - 1) / GBN);
    if (Nd % GBN == 0)
      gemm_bf16_wmma<true><<<grid, 256, 0, stream>>>(A, Bm, Cp, M, Nd, Kd);
    else
      gemm_bf16_wmma<false><<<grid, 256, 0, stream>>>(A, Bm, Cp, M, Nd, Kd);
  };

  // 1) weights -> bf16
  cvt(Wr, Wr_bf, CC, 0);  cvt(Wk, Wk_bf, CC, 0);
  cvt(Wv, Wv_bf, CC, 0);  cvt(Wo, Wo_bf, CC, 0);
  cvt(w1, w1_bf, (size_t)Cq * DWq, 0);  cvt(w2, w2_bf, (size_t)DWq * Cq, 0);
  cvt(v1, v1_bf, (size_t)Cq * DVq, 0);  cvt(v2, v2_bf, (size_t)DVq * Cq, 0);
  cvt(a1, a1_bf, (size_t)Cq * DAq, 0);  cvt(a2, a2_bf, (size_t)DAq * Cq, 0);
  cvt(g1, g1_bf, (size_t)Cq * DGq, 0);  cvt(g2, g2_bf, (size_t)DGq * Cq, 0);
  cvt(WkF, WkF_bf, (size_t)Cq * DFFq, 0);
  cvt(WvF, WvF_bf, (size_t)DFFq * Cq, 0);

  // 2) LN1 + token-shift mixes
  ln_kernel<<<MT, 256, 0, stream>>>(x, ln1_w, ln1_b, xn);
  mix6_kernel<<<(int)((BTC + 255) / 256), 256, 0, stream>>>(
      xn, att_shift, x_r, x_w, x_k, x_v, x_a, x_g, xr, xw, xkb, xv, xa, xg);

  // 3) projections (WMMA)
  gemm(xr, Wr_bf, rbuf, MT, Cq, Cq);
  gemm(xkb, Wk_bf, kbuf, MT, Cq, Cq);
  gemm(xv, Wv_bf, vbuf, MT, Cq, Cq);
  gemm(xw, w1_bf, hw, MT, DWq, Cq);
  gemm(xv, v1_bf, hv, MT, DVq, Cq);
  gemm(xa, a1_bf, ha, MT, DAq, Cq);
  gemm(xg, g1_bf, hg, MT, DGq, Cq);

  // 4) LoRA inner activations + second matmuls
  cvt(hw, hw_bf, (size_t)MT * DWq, 1);   // tanh
  cvt(hv, hv_bf, (size_t)MT * DVq, 0);
  cvt(ha, ha_bf, (size_t)MT * DAq, 0);
  cvt(hg, hg_bf, (size_t)MT * DGq, 2);   // sigmoid
  gemm(hw_bf, w2_bf, wlin, MT, Cq, DWq);
  gemm(hv_bf, v2_bf, vlin, MT, Cq, DVq);
  gemm(ha_bf, a2_bf, alin, MT, Cq, DAq);
  gemm(hg_bf, g2_bf, gbuf, MT, Cq, DGq);

  // 5) WKV prep + sequential scan (final state -> d_out slice)
  prep_wkv_kernel<<<MT * Hq, 64, 0, stream>>>(
      kbuf, wlin, vbuf, vlin, alin, v_first, w0, v0, a0, k_k, k_a,
      wh, kh, vh, aab, bbb);
  wkv_scan_kernel<<<Bq * Hq, 256, 0, stream>>>(
      rbuf, wh, kh, vh, aab, bbb, wkv_state, ybuf, out_S);

  // 6) GroupNorm + residual + gate, output projection, residual add
  gn_resid_gate_kernel<<<MT * Hq, 64, 0, stream>>>(
      ybuf, rbuf, kh, vh, gbuf, r_k, lnx_w, lnx_b, og);
  gemm(og, Wo_bf, att, MT, Cq, Cq);
  add_kernel<<<16384, 256, 0, stream>>>(x, att, x1, BTC);

  // 7) ChannelMix
  ln_kernel<<<MT, 256, 0, stream>>>(x1, ln2_w, ln2_b, xn2);
  mix1_kernel<<<(int)((BTC + 255) / 256), 256, 0, stream>>>(xn2, ffn_shift, ffn_xk, xkf);
  gemm(xkf, WkF_bf, hffn, MT, DFFq, Cq);
  cvt(hffn, kf, (size_t)MT * DFFq, 3);   // relu^2
  gemm(kf, WvF_bf, ffnout, MT, Cq, DFFq);
  add_kernel<<<16384, 256, 0, stream>>>(x1, ffnout, out_x, BTC);
}
